// TorusNeuralSDE_52020643889620
// MI455X (gfx1250) — matrix operand order repaired and checked by reading, not compiled, not run on hardware
//
#include <hip/hip_runtime.h>
#include <hip/hip_bf16.h>

// ---------------------------------------------------------------------------
// TorusNeuralSDE on gfx1250: WMMA GRU scan (async double-buffered input,
// TDM-staged recurrent weights) + WMMA SDE MLP integrator.
// ---------------------------------------------------------------------------

#define B_TOT   1024
#define T_LEN   256
#define Dd      256
#define Hh      128
#define Cc      64
#define D2      512          // 2d
#define G3      384          // 3H
#define IN_SDE  576          // 2d + C
#define BT      32           // batch rows per workgroup
#define NTHR    128          // 4 waves of 32
#define PI_F        3.14159265358979323846f
#define TWO_PI_F    6.28318530717958647692f
#define INV_2PI_F   0.15915494309189533577f
#define DT_F        0.05f
#define DIFF_SCALE_F 0.3f

typedef __attribute__((ext_vector_type(16))) _Float16 v16h;
typedef __attribute__((ext_vector_type(8)))  _Float16 v8h;
typedef __attribute__((ext_vector_type(8)))  float    v8f;
typedef __attribute__((ext_vector_type(4)))  unsigned int u32x4;
typedef __attribute__((ext_vector_type(8)))  int          i32x8;
typedef __attribute__((ext_vector_type(4)))  int          i32x4;

// ---- CDNA5 feature guards (degrade gracefully if builtin missing) ----
#if defined(__AMDGCN__) && __has_builtin(__builtin_amdgcn_global_load_async_to_lds_b128)
#define HAS_ASYNC_LDS 1
#else
#define HAS_ASYNC_LDS 0
#endif

#if defined(__AMDGCN__) && __has_builtin(__builtin_amdgcn_tensor_load_to_lds) && \
    __has_builtin(__builtin_amdgcn_s_wait_tensorcnt)
#define HAS_TDM 1
#else
#define HAS_TDM 0
#endif

#if HAS_ASYNC_LDS
#if __has_builtin(__builtin_amdgcn_s_wait_asynccnt)
#define WAIT_ASYNC() __builtin_amdgcn_s_wait_asynccnt(0)
#else
#define WAIT_ASYNC() asm volatile("s_wait_asynccnt 0x0" ::: "memory")
#endif
#else
#define WAIT_ASYNC()
#endif

// ---- workspace byte offsets (all 1 KiB aligned) ----
#define OFF_WIH   0u         // 24nt x 16kt x 1024B = 393216
#define OFF_WHH   393216u    // 24 x 4  x 1024      =  98304
#define OFF_PROJ  491520u    //  4 x 4  x 1024      =  16384
#define OFF_DR0   507904u    //  8 x 18 x 1024      = 147456
#define OFF_DR1   655360u    //  8 x 4  x 1024      =  32768
#define OFF_DR2   688128u
#define OFF_DR3   720896u    // 16 x 4  x 1024      =  65536
#define OFF_DF0   786432u    // 147456
#define OFF_DF1   933888u
#define OFF_DF2   966656u    //  65536
#define OFF_CTX   1032192u   // B*C f32             = 262144
// total: 1294336 bytes (~1.24 MB)

#define WHH_DWORDS 24576     // 98304 B of swizzled W_hh tiles

// ---------------------------------------------------------------------------
// math helpers
// ---------------------------------------------------------------------------
__device__ __forceinline__ float fsigmoid(float x) {
    return 1.0f / (1.0f + __expf(-x));
}
__device__ __forceinline__ float ftanh(float x) {
    float e = __expf(-2.0f * x);
    return (1.0f - e) / (1.0f + e);
}
__device__ __forceinline__ float fsoftplus(float x) {
    return (x > 20.0f) ? x : __logf(1.0f + __expf(x));
}
__device__ __forceinline__ float wrap_pi(float x) {
    float y = (x + PI_F) * INV_2PI_F;
    y = y - floorf(y);
    return y * TWO_PI_F - PI_F;
}

// ---------------------------------------------------------------------------
// WMMA fragment helpers (CDNA5 16-bit layouts, wave32)
// ---------------------------------------------------------------------------
__device__ __forceinline__ v8f wmma16(v16h a, v16h b, v8f c) {
    return __builtin_amdgcn_wmma_f32_16x16x32_f16(
        false, a, false, b, (short)0, c, false, false);
}

// A-matrix 16x32 f16 (ISA 7.12.2): lanes 0-15 K {0-7,16-23}, lanes 16-31 K {8-15,24-31}
__device__ __forceinline__ v16h load_a_frag(const _Float16* __restrict__ base,
                                            int ldk, int mt, int kt, int lane) {
    int m  = mt * 16 + (lane & 15);
    int kh = ((lane >> 4) & 1) * 8;
    const _Float16* p = base + (size_t)m * ldk + kt * 32 + kh;
    union { v16h v; v8h h[2]; } u;
    u.h[0] = *(const v8h*)p;
    u.h[1] = *(const v8h*)(p + 16);
    return u.v;
}

// B-matrix 32x16 f16, pre-swizzled: lane's 16 halves contiguous (32 B / lane)
__device__ __forceinline__ v16h load_b_frag(const _Float16* __restrict__ tile,
                                            int lane) {
    const _Float16* p = tile + lane * 16;
    union { v16h v; v8h h[2]; } u;
    u.h[0] = *(const v8h*)p;
    u.h[1] = *(const v8h*)(p + 8);
    return u.v;
}

// C/D 16x16 f32: VGPR r, lanes 0-15 -> M=r, lanes 16-31 -> M=8+r, N=lane&15
__device__ __forceinline__ void store_c(float* __restrict__ base, int ldn,
                                        int mt, int nt, int lane, v8f acc) {
    int n    = nt * 16 + (lane & 15);
    int mrow = mt * 16 + ((lane >> 4) & 1) * 8;
#pragma unroll
    for (int r = 0; r < 8; ++r) base[(mrow + r) * ldn + n] = acc[r];
}

// Generic 32xN GEMM over a workgroup (wave&1 -> M-tile, wave>>1 -> N half)
template <bool SILU, bool F16OUT>
__device__ __forceinline__ void gemm_block(const _Float16* __restrict__ inp, int ldk,
                                           int nKt,
                                           const _Float16* __restrict__ wsw,
                                           const float* __restrict__ bias,
                                           void* __restrict__ outv, int ldn, int nNt,
                                           int wave, int lane) {
    int mt    = wave & 1;
    int ntPer = nNt >> 1;
    int nt0   = (wave >> 1) * ntPer;
    for (int i = 0; i < ntPer; ++i) {
        int nt = nt0 + i;
        const _Float16* wt = wsw + (size_t)nt * nKt * 512;
        v8f acc = {};
        for (int kt = 0; kt < nKt; ++kt) {
            __builtin_prefetch(wt + (kt + 1) * 512, 0, 0);  // global_prefetch_b8
            v16h a = load_a_frag(inp, ldk, mt, kt, lane);
            v16h b = load_b_frag(wt + kt * 512, lane);
            acc = wmma16(a, b, acc);
        }
        int   n    = nt * 16 + (lane & 15);
        int   mrow = mt * 16 + ((lane >> 4) & 1) * 8;
        float bb   = bias[n];
#pragma unroll
        for (int r = 0; r < 8; ++r) {
            float v = acc[r] + bb;
            if (SILU) v = v * fsigmoid(v);
            if (F16OUT) ((_Float16*)outv)[(mrow + r) * ldn + n] = (_Float16)v;
            else        ((float*)outv)[(mrow + r) * ldn + n]     = v;
        }
    }
}

// ---------------------------------------------------------------------------
// Async global -> LDS staging of one thread's 64-float contiguous chunk
// (16 x global_load_async_to_lds_b128, tracked by ASYNCcnt)
// ---------------------------------------------------------------------------
__device__ __forceinline__ void stage_chunk(const float* __restrict__ g,
                                            float* __restrict__ l) {
#if HAS_ASYNC_LDS
#pragma unroll
    for (int q = 0; q < 16; ++q)
        __builtin_amdgcn_global_load_async_to_lds_b128(
            (__attribute__((address_space(1))) i32x4*)(g + q * 4),
            (__attribute__((address_space(3))) i32x4*)(l + q * 4), 0, 0);
#else
#pragma unroll
    for (int q = 0; q < 16; ++q)
        *(float4*)(l + q * 4) = *(const float4*)(g + q * 4);
#endif
}

// ---------------------------------------------------------------------------
// One-shot TDM linear DMA (global -> LDS) of ndw dwords, per ISA D# layout.
// ---------------------------------------------------------------------------
#if HAS_TDM
__device__ __forceinline__ void tdm_load_linear(unsigned lds_addr,
                                                const void* gptr,
                                                unsigned ndw) {
    unsigned long long ga = (unsigned long long)gptr;
    u32x4 g0 = {};
    g0[0] = 1u;                                            // count=1 (valid user D#)
    g0[1] = lds_addr;                                      // lds_addr
    g0[2] = (unsigned)ga;                                  // global_addr[31:0]
    g0[3] = (unsigned)((ga >> 32) & 0x01ffffffu) | (2u << 30);  // addr[56:32] | type=2
    i32x8 g1 = {};
    g1[0] = (int)(2u << 16);                               // data_size=2 (4B), mask=0
    g1[1] = (int)((ndw & 0xffffu) << 16);                  // tensor_dim0[15:0]
    g1[2] = (int)((ndw >> 16) & 0xffffu) | (1 << 16);      // tensor_dim0[31:16], tensor_dim1=1
    g1[3] = (int)((ndw & 0xffffu) << 16);                  // tile_dim0 = ndw (16-bit)
    g1[4] = 1;                                             // tile_dim1 = 1
    g1[5] = (int)ndw;                                      // tensor_dim0_stride[31:0]
    i32x4 g2 = {};
    i32x4 g3 = {};
#if defined(__clang_major__) && (__clang_major__ >= 23)
    i32x8 g4 = {};
    __builtin_amdgcn_tensor_load_to_lds(g0, g1, g2, g3, g4, 0);
#else
    __builtin_amdgcn_tensor_load_to_lds(g0, g1, g2, g3, 0);
#endif
}
#endif

// ---------------------------------------------------------------------------
// Kernel 0: convert fp32 weight W[N][K] into swizzled f16 WMMA-B tiles.
// ---------------------------------------------------------------------------
__global__ void swizzle_b_kernel(const float* __restrict__ W,
                                 _Float16* __restrict__ sw, int N, int K) {
    int nKt = K >> 5;
    int idx = blockIdx.x * 256 + threadIdx.x;
    if (idx >= N * K) return;
    int e    = idx & 15;
    int l    = (idx >> 4) & 31;
    int tile = idx >> 9;
    int kt   = tile % nKt;
    int nt   = tile / nKt;
    int n    = nt * 16 + (l & 15);
    int k    = kt * 32 + ((l >> 4) & 1) * 8 + (e & 7) + (e >> 3) * 16;
    sw[idx] = (_Float16)W[(size_t)n * K + k];
}

// ---------------------------------------------------------------------------
// Kernel 1: persistent GRU scan.
//  - angle tiles double-buffered via async global->LDS DMA (ASYNCcnt)
//  - W_hh swizzled tiles staged once into LDS via TDM (TENSORcnt)
//  - all GEMMs on v_wmma_f32_16x16x32_f16; h persists in LDS across 256 steps
// LDS: 319488 B of the 320 KB WGP pool (one WG per WGP).
// ---------------------------------------------------------------------------
__global__ void gru_kernel(const float* __restrict__ ang,
                           const _Float16* __restrict__ wih_sw,
                           const _Float16* __restrict__ whh_sw,
                           const float* __restrict__ gb,
                           const float* __restrict__ gbn,
                           const _Float16* __restrict__ proj_sw,
                           const float* __restrict__ pb,
                           float* __restrict__ ctx_out) {
    extern __shared__ char smem[];
    _Float16* feats = (_Float16*)smem;              // 32x512 f16 = 32768 B
    _Float16* hf16  = (_Float16*)(smem + 32768);    // 32x128 f16 =  8192 B
    float*    hf32  = (float*)(smem + 40960);       // 32x128 f32 = 16384 B
    float*    gx    = (float*)(smem + 57344);       // 32x384 f32 = 49152 B
    float*    gh    = (float*)(smem + 106496);      // 32x384 f32 = 49152 B
    float*    angb  = (float*)(smem + 155648);      // 2 x 32x256 f32 = 65536 B
    _Float16* whhL  = (_Float16*)(smem + 221184);   // swizzled W_hh  = 98304 B

    const int tid  = threadIdx.x;
    const int lane = tid & 31;
    const int wave = tid >> 5;
    const int b0   = blockIdx.x * BT;

    // this thread's contiguous 64-float chunk of the (32 x 256) angle tile
    const int cr = tid >> 2;
    const int cj = (tid & 3) * 64;
    const float* gsrc = ang + ((size_t)(b0 + cr) * T_LEN) * Dd + cj;  // +t*Dd/step
    const int    loff = cr * Dd + cj;

    for (int i = tid; i < BT * Hh; i += NTHR) {
        hf32[i] = 0.0f;
        hf16[i] = (_Float16)0.0f;
    }

    // stage W_hh tiles into LDS once (TDM DMA, fallback: cooperative copy)
#if HAS_TDM
    if (wave == 0 && lane == 0) {
        unsigned lds_addr =
            (unsigned)(size_t)(__attribute__((address_space(3))) void*)whhL;
        tdm_load_linear(lds_addr, whh_sw, WHH_DWORDS);
    }
    if (wave == 0) __builtin_amdgcn_s_wait_tensorcnt(0);
#else
    {
        const unsigned* s = (const unsigned*)whh_sw;
        unsigned*       d = (unsigned*)whhL;
        for (int i = tid; i < WHH_DWORDS; i += NTHR) d[i] = s[i];
    }
#endif

    // prologue: stage angles for t = 0 into buffer 0
    stage_chunk(gsrc, angb + loff);
    WAIT_ASYNC();
    __syncthreads();

    const int mt  = wave & 1;
    const int nt0 = (wave >> 1) * 12;

    for (int t = 0; t < T_LEN; ++t) {
        const float* cur = (t & 1) ? (angb + 8192) : angb;

        // sin/cos features for this timestep -> f16 LDS
        for (int i = tid; i < BT * Dd; i += NTHR) {
            int r = i >> 8, j = i & 255;
            float s, c;
            __sincosf(cur[i], &s, &c);
            feats[r * D2 + j]      = (_Float16)s;
            feats[r * D2 + Dd + j] = (_Float16)c;
        }
        __syncthreads();

        // kick off async DMA of the NEXT timestep's angles (overlaps WMMAs)
        if (t + 1 < T_LEN) {
            float* nxt = ((t + 1) & 1) ? (angb + 8192) : angb;
            stage_chunk(gsrc + (size_t)(t + 1) * Dd, nxt + loff);
        }

        // gx = x @ Wih^T + b ; gh = h @ Whh^T (separate: n-gate needs them apart)
        for (int i = 0; i < 12; ++i) {
            int nt = nt0 + i;
            v8f acc = {};
            for (int kt = 0; kt < 16; ++kt) {   // K = 512, B from L2
                v16h a = load_a_frag(feats, D2, mt, kt, lane);
                v16h b = load_b_frag(wih_sw + ((size_t)nt * 16 + kt) * 512, lane);
                acc = wmma16(a, b, acc);
            }
            float bb = gb[nt * 16 + (lane & 15)];
#pragma unroll
            for (int r = 0; r < 8; ++r) acc[r] += bb;
            store_c(gx, G3, mt, nt, lane, acc);

            v8f acc2 = {};
            for (int kt = 0; kt < 4; ++kt) {    // K = 128, B from LDS (TDM-staged)
                v16h a = load_a_frag(hf16, Hh, mt, kt, lane);
                v16h b = load_b_frag(whhL + ((size_t)nt * 4 + kt) * 512, lane);
                acc2 = wmma16(a, b, acc2);
            }
            store_c(gh, G3, mt, nt, lane, acc2);
        }
        __syncthreads();

        // gate math + hidden update
        for (int i = tid; i < BT * Hh; i += NTHR) {
            int r = i >> 7, n = i & 127;
            const float* gxr = gx + r * G3;
            const float* ghr = gh + r * G3;
            float rr = fsigmoid(gxr[n] + ghr[n]);
            float zz = fsigmoid(gxr[Hh + n] + ghr[Hh + n]);
            float nn = ftanh(gxr[2 * Hh + n] + rr * (ghr[2 * Hh + n] + gbn[n]));
            float h  = hf32[i];
            float hN = nn + zz * (h - nn);
            hf32[i] = hN;
            hf16[i] = (_Float16)hN;
        }
        WAIT_ASYNC();          // next-step angle DMA must be in LDS
        __syncthreads();
    }

    // ctx = h @ proj_w^T + proj_b : 2 M-tiles x 4 N-tiles, K = 128
    for (int q = wave; q < 8; q += 4) {
        int mtq = q >> 2, ntq = q & 3;
        v8f acc = {};
        for (int kt = 0; kt < 4; ++kt) {
            v16h a = load_a_frag(hf16, Hh, mtq, kt, lane);
            v16h b = load_b_frag(proj_sw + ((size_t)ntq * 4 + kt) * 512, lane);
            acc = wmma16(a, b, acc);
        }
        int   n    = ntq * 16 + (lane & 15);
        int   mrow = mtq * 16 + ((lane >> 4) & 1) * 8;
        float bb   = pb[n];
#pragma unroll
        for (int r = 0; r < 8; ++r)
            ctx_out[(size_t)(b0 + mrow + r) * Cc + n] = acc[r] + bb;
    }
}

// ---------------------------------------------------------------------------
// Kernel 2: SDE integrator. theta tile persists in LDS across the 5 steps;
// 7 WMMA GEMM layers per step with fused silu epilogues (softplus applied in the theta update).
// ---------------------------------------------------------------------------
__global__ void sde_kernel(const float* __restrict__ ang,
                           const float* __restrict__ dW,
                           const float* __restrict__ ctx_in,
                           const _Float16* dr0, const float* drb0,
                           const _Float16* dr1, const float* drb1,
                           const _Float16* dr2, const float* drb2,
                           const _Float16* dr3, const float* drb3,
                           const _Float16* df0, const float* dfb0,
                           const _Float16* df1, const float* dfb1,
                           const _Float16* df2, const float* dfb2,
                           float* __restrict__ out) {
    extern __shared__ char smem[];
    float*    theta = (float*)smem;                 // 32x256 f32 = 32768 B
    _Float16* inpF  = (_Float16*)(smem + 32768);    // 32x576 f16 = 36864 B
    _Float16* actA  = (_Float16*)(smem + 69632);    // 32x128 f16 =  8192 B
    _Float16* actB  = (_Float16*)(smem + 77824);    // 32x128 f16 =  8192 B
    float*    mu    = (float*)(smem + 86016);       // 32x256 f32 = 32768 B
    float*    sg    = (float*)(smem + 118784);      // 32x256 f32 = 32768 B

    const int tid  = threadIdx.x;
    const int lane = tid & 31;
    const int wave = tid >> 5;
    const int b0   = blockIdx.x * BT;

    for (int i = tid; i < BT * Dd; i += NTHR) {
        int r = i >> 8, j = i & 255;
        theta[i] = wrap_pi(ang[((size_t)(b0 + r) * T_LEN + (T_LEN - 1)) * Dd + j]);
    }
    for (int i = tid; i < BT * Cc; i += NTHR) {
        int r = i >> 6, c = i & 63;
        inpF[r * IN_SDE + D2 + c] = (_Float16)ctx_in[(size_t)(b0 + r) * Cc + c];
    }
    __syncthreads();

    for (int k = 0; k < 5; ++k) {
        for (int i = tid; i < BT * Dd; i += NTHR) {
            int r = i >> 8, j = i & 255;
            float s, c;
            __sincosf(theta[i], &s, &c);
            inpF[r * IN_SDE + j]      = (_Float16)s;
            inpF[r * IN_SDE + Dd + j] = (_Float16)c;
        }
        __syncthreads();

        // drift MLP: 576 -> 128 -> 128 -> 128 -> 256
        gemm_block<true,  true >(inpF, IN_SDE, 18, dr0, drb0, actA, Hh, 8,  wave, lane);
        __syncthreads();
        gemm_block<true,  true >(actA, Hh,     4,  dr1, drb1, actB, Hh, 8,  wave, lane);
        __syncthreads();
        gemm_block<true,  true >(actB, Hh,     4,  dr2, drb2, actA, Hh, 8,  wave, lane);
        __syncthreads();
        gemm_block<false, false>(actA, Hh,     4,  dr3, drb3, mu,   Dd, 16, wave, lane);
        __syncthreads();

        // diffusion MLP: 576 -> 128 -> 128 -> 256 (pre-softplus)
        gemm_block<true,  true >(inpF, IN_SDE, 18, df0, dfb0, actB, Hh, 8,  wave, lane);
        __syncthreads();
        gemm_block<true,  true >(actB, Hh,     4,  df1, dfb1, actA, Hh, 8,  wave, lane);
        __syncthreads();
        gemm_block<false, false>(actA, Hh,     4,  df2, dfb2, sg,   Dd, 16, wave, lane);
        __syncthreads();

        // theta update on the flat torus
        for (int i = tid; i < BT * Dd; i += NTHR) {
            int r = i >> 8, j = i & 255;
            float s  = fsoftplus(sg[i]) * DIFF_SCALE_F;
            float dw = dW[((size_t)(b0 + r) * 5 + k) * Dd + j];
            theta[i] = wrap_pi(theta[i] + mu[i] * DT_F + s * dw);
        }
        __syncthreads();
    }

    for (int i = tid; i < BT * Dd; i += NTHR) {
        int r = i >> 8, j = i & 255;
        out[(size_t)(b0 + r) * Dd + j] = theta[i];
    }
}

// ---------------------------------------------------------------------------
// host entry
// ---------------------------------------------------------------------------
extern "C" void kernel_launch(void* const* d_in, const int* in_sizes, int n_in,
                              void* d_out, int out_size, void* d_ws, size_t ws_size,
                              hipStream_t stream) {
    (void)in_sizes; (void)n_in; (void)out_size; (void)ws_size;

    const float* ang = (const float*)d_in[0];
    const float* dW  = (const float*)d_in[1];

    char* ws = (char*)d_ws;
    _Float16* wih  = (_Float16*)(ws + OFF_WIH);
    _Float16* whh  = (_Float16*)(ws + OFF_WHH);
    _Float16* proj = (_Float16*)(ws + OFF_PROJ);
    _Float16* dr0  = (_Float16*)(ws + OFF_DR0);
    _Float16* dr1  = (_Float16*)(ws + OFF_DR1);
    _Float16* dr2  = (_Float16*)(ws + OFF_DR2);
    _Float16* dr3  = (_Float16*)(ws + OFF_DR3);
    _Float16* df0  = (_Float16*)(ws + OFF_DF0);
    _Float16* df1  = (_Float16*)(ws + OFF_DF1);
    _Float16* df2  = (_Float16*)(ws + OFF_DF2);
    float*    ctxb = (float*)(ws + OFF_CTX);

    auto sw = [&](const void* W, _Float16* dst, int N, int K) {
        int total = N * K;
        swizzle_b_kernel<<<(total + 255) / 256, 256, 0, stream>>>(
            (const float*)W, dst, N, K);
    };
    sw(d_in[2],  wih,  G3, D2);
    sw(d_in[3],  whh,  G3, Hh);
    sw(d_in[6],  proj, Cc, Hh);
    sw(d_in[8],  dr0,  Hh, IN_SDE);
    sw(d_in[10], dr1,  Hh, Hh);
    sw(d_in[12], dr2,  Hh, Hh);
    sw(d_in[14], dr3,  Dd, Hh);
    sw(d_in[16], df0,  Hh, IN_SDE);
    sw(d_in[18], df1,  Hh, Hh);
    sw(d_in[20], df2,  Dd, Hh);

    gru_kernel<<<B_TOT / BT, NTHR, 319488, stream>>>(
        ang, wih, whh,
        (const float*)d_in[4],
        (const float*)d_in[5],
        proj,
        (const float*)d_in[7],
        ctxb);

    sde_kernel<<<B_TOT / BT, NTHR, 151552, stream>>>(
        ang, dW, ctxb,
        dr0, (const float*)d_in[9],
        dr1, (const float*)d_in[11],
        dr2, (const float*)d_in[13],
        dr3, (const float*)d_in[15],
        df0, (const float*)d_in[17],
        df1, (const float*)d_in[19],
        df2, (const float*)d_in[21],
        (float*)d_out);
}
